// RealMambaBlock_29729763623372
// MI455X (gfx1250) — matrix operand order repaired and checked by reading, not compiled, not run on hardware
//
#include <hip/hip_runtime.h>

typedef __attribute__((ext_vector_type(16))) _Float16 v16h;
typedef __attribute__((ext_vector_type(8)))  _Float16 v8h;
typedef __attribute__((ext_vector_type(8)))  float    v8f;

#define B_SZ     2
#define L_SZ     2048
#define D_MODEL  1024
#define D_INNER  2048
#define D_STATE  16
#define DT_RANK  64
#define NROWS    (B_SZ * L_SZ)          // 4096
#define NXZ      (2 * D_INNER)          // 4096
#define NXP      (DT_RANK + 2*D_STATE)  // 96

// ---------------- LayerNorm -> f16 activations ----------------
__global__ void k_layernorm(const float* __restrict__ x, const float* __restrict__ g,
                            const float* __restrict__ b, _Float16* __restrict__ out) {
  const int row = blockIdx.x;
  const float* xr = x + (size_t)row * D_MODEL;
  __shared__ float s1[256], s2[256];
  float sum = 0.f, sq = 0.f;
  for (int i = threadIdx.x; i < D_MODEL; i += 256) { float v = xr[i]; sum += v; sq += v * v; }
  s1[threadIdx.x] = sum; s2[threadIdx.x] = sq; __syncthreads();
  for (int s = 128; s > 0; s >>= 1) {
    if ((int)threadIdx.x < s) { s1[threadIdx.x] += s1[threadIdx.x + s]; s2[threadIdx.x] += s2[threadIdx.x + s]; }
    __syncthreads();
  }
  const float mu   = s1[0] * (1.f / D_MODEL);
  const float var  = s2[0] * (1.f / D_MODEL) - mu * mu;
  const float rstd = rsqrtf(var + 1e-5f);
  _Float16* orow = out + (size_t)row * D_MODEL;
  for (int i = threadIdx.x; i < D_MODEL; i += 256)
    orow[i] = (_Float16)((xr[i] - mu) * rstd * g[i] + b[i]);
}

// ---------------- W [K,N] f32 -> Wt [N,K] f16 ----------------
__global__ void k_transpose_f16(const float* __restrict__ W, _Float16* __restrict__ Wt,
                                int K, int N) {
  size_t idx = (size_t)blockIdx.x * 256 + threadIdx.x;
  if (idx >= (size_t)K * N) return;
  int k = (int)(idx % K);
  int n = (int)(idx / K);
  Wt[idx] = (_Float16)W[(size_t)k * N + n];
}

// ---------------- generic f16 WMMA GEMM, f32 accumulate ----------------
// C[M,N] = A[M,K] * Bt[N,K]^T ; modes: 0 plain, 1 softplus(acc+bias[n]), 2 acc+resid
// Each wave owns a 32x32 output tile (2x2 WMMA accumulators) so every A/B
// fragment load feeds two v_wmma ops: 16 FLOP/byte register-level intensity.
// Block = 8 waves in a 2x4 grid -> 64x128 block tile.
__global__ void k_gemm_wmma(const _Float16* __restrict__ A, const _Float16* __restrict__ Bt,
                            float* __restrict__ C, int M, int N, int K,
                            int mode, const float* __restrict__ bias,
                            const float* __restrict__ resid) {
  const int lane  = threadIdx.x & 31;
  const int warp  = threadIdx.x >> 5;
  const int tileM = blockIdx.y * 64  + (warp >> 2) * 32;
  const int tileN = blockIdx.x * 128 + (warp & 3) * 32;
  if (tileM >= M || tileN >= N) return;   // whole wave exits; EXEC all-ones otherwise

  const int m16 = lane & 15;
  const int hi  = lane >> 4;              // 0 for lanes 0-15, 1 for lanes 16-31
  // A fragment: lane holds row; halves 0..7 = K [kb,kb+8), 8..15 = K [kb+16,kb+24), kb = 8*hi
  const _Float16* ap0 = A  + (size_t)(tileM + m16) * K + hi * 8;
  const _Float16* ap1 = ap0 + (size_t)16 * K;
  // B fragment: lane holds col; halves 0..7 = K [kb,kb+8), 8..15 = K [kb+16,kb+24), kb = 8*hi
  const _Float16* bp0 = Bt + (size_t)(tileN + m16) * K + hi * 8;
  const _Float16* bp1 = bp0 + (size_t)16 * K;

  v8f acc[2][2] = {};
  union F { v16h v; v8h h[2]; };
  for (int k0 = 0; k0 < K; k0 += 32) {
    F a0, a1, b0, b1;
    a0.h[0] = *(const v8h*)(ap0 + k0); a0.h[1] = *(const v8h*)(ap0 + k0 + 16);
    a1.h[0] = *(const v8h*)(ap1 + k0); a1.h[1] = *(const v8h*)(ap1 + k0 + 16);
    b0.h[0] = *(const v8h*)(bp0 + k0); b0.h[1] = *(const v8h*)(bp0 + k0 + 16);
    b1.h[0] = *(const v8h*)(bp1 + k0); b1.h[1] = *(const v8h*)(bp1 + k0 + 16);
    // prefetch the streamed operands a few cachelines ahead (global_prefetch_b8)
    __builtin_prefetch(ap0 + k0 + 256, 0, 1);
    __builtin_prefetch(ap1 + k0 + 256, 0, 1);
    __builtin_prefetch(bp0 + k0 + 256, 0, 1);
    __builtin_prefetch(bp1 + k0 + 256, 0, 1);
    acc[0][0] = __builtin_amdgcn_wmma_f32_16x16x32_f16(false, a0.v, false, b0.v,
                                                       (short)0, acc[0][0], false, false);
    acc[0][1] = __builtin_amdgcn_wmma_f32_16x16x32_f16(false, a0.v, false, b1.v,
                                                       (short)0, acc[0][1], false, false);
    acc[1][0] = __builtin_amdgcn_wmma_f32_16x16x32_f16(false, a1.v, false, b0.v,
                                                       (short)0, acc[1][0], false, false);
    acc[1][1] = __builtin_amdgcn_wmma_f32_16x16x32_f16(false, a1.v, false, b1.v,
                                                       (short)0, acc[1][1], false, false);
  }

  // C/D layout: VGPR v -> row sub-tile base + 8*hi + v, col sub-tile base + m16
  #pragma unroll
  for (int i = 0; i < 2; ++i) {
    #pragma unroll
    for (int j = 0; j < 2; ++j) {
      const int rbase = tileM + i * 16 + (hi << 3);
      const int cidx  = tileN + j * 16 + m16;
      #pragma unroll
      for (int v = 0; v < 8; ++v) {
        float val = acc[i][j][v];
        if (mode == 1) {
          val += bias[cidx];
          val = (val > 20.f) ? val : log1pf(__expf(val));   // softplus
        } else if (mode == 2) {
          val += resid[(size_t)(rbase + v) * N + cidx];     // residual add
        }
        C[(size_t)(rbase + v) * N + cidx] = val;
      }
    }
  }
}

// ---------------- causal depthwise conv (4 taps) + bias + SiLU ----------------
__global__ void k_conv_silu(const float* __restrict__ xz, const float* __restrict__ cw,
                            const float* __restrict__ cb, float* __restrict__ u32,
                            _Float16* __restrict__ u16) {
  size_t idx = (size_t)blockIdx.x * 256 + threadIdx.x;   // < NROWS * D_INNER
  const int d   = (int)(idx & (D_INNER - 1));
  const int row = (int)(idx >> 11);
  const int l   = row & (L_SZ - 1);
  const int b   = row >> 11;
  float acc = cb[d];
  #pragma unroll
  for (int j = 0; j < 4; ++j) {
    int ll = l - 3 + j;
    if (ll >= 0)
      acc += cw[d * 4 + j] * xz[((size_t)(b * L_SZ + ll)) * NXZ + d];   // u = first half of xz
  }
  float s = acc / (1.f + __expf(-acc));   // SiLU
  u32[idx] = s;
  u16[idx] = (_Float16)s;
}

// ---------------- slice dt-rank part of x_dbl -> f16 ----------------
__global__ void k_slice_f16(const float* __restrict__ xdbl, _Float16* __restrict__ dtp) {
  int idx = blockIdx.x * 256 + threadIdx.x;   // < NROWS * DT_RANK
  int c = idx & (DT_RANK - 1);
  int r = idx >> 6;
  dtp[idx] = (_Float16)xdbl[(size_t)r * NXP + c];
}

// ---------------- selective scan + skip + gate, emits y in f16 ----------------
__global__ void k_scan(const float* __restrict__ dt, const float* __restrict__ u32,
                       const float* __restrict__ xdbl, const float* __restrict__ xz,
                       const float* __restrict__ A_log, const float* __restrict__ Dp,
                       _Float16* __restrict__ y16) {
  const int d = blockIdx.x * 256 + threadIdx.x;   // channel 0..D_INNER-1
  const int b = blockIdx.y;
  __shared__ float sB[D_STATE], sC[D_STATE];
  float Areg[D_STATE], st[D_STATE];
  #pragma unroll
  for (int n = 0; n < D_STATE; ++n) {
    Areg[n] = -__expf(A_log[(size_t)d * D_STATE + n]);
    st[n] = 0.f;
  }
  const float Dd = Dp[d];
  for (int l = 0; l < L_SZ; ++l) {
    const size_t row = (size_t)b * L_SZ + l;
    if (threadIdx.x < 32) {
      float v = xdbl[row * NXP + DT_RANK + threadIdx.x];
      if (threadIdx.x < 16) sB[threadIdx.x] = v; else sC[threadIdx.x - 16] = v;
    }
    __syncthreads();
    const float dtv = dt[row * D_INNER + d];
    const float uv  = u32[row * D_INNER + d];
    const float du  = dtv * uv;
    float y = 0.f;
    #pragma unroll
    for (int n = 0; n < D_STATE; ++n) {
      float dA = __expf(dtv * Areg[n]);
      st[n] = dA * st[n] + du * sB[n];
      y += st[n] * sC[n];
    }
    y += uv * Dd;                                   // skip connection
    const float zv = xz[row * NXZ + D_INNER + d];   // z = second half of xz
    y *= zv / (1.f + __expf(-zv));                  // gate with SiLU(z)
    y16[row * D_INNER + d] = (_Float16)y;
    __syncthreads();
  }
}

extern "C" void kernel_launch(void* const* d_in, const int* in_sizes, int n_in,
                              void* d_out, int out_size, void* d_ws, size_t ws_size,
                              hipStream_t stream) {
  (void)in_sizes; (void)n_in; (void)out_size; (void)ws_size;
  const float* x      = (const float*)d_in[0];
  const float* ln_g   = (const float*)d_in[1];
  const float* ln_b   = (const float*)d_in[2];
  const float* W_in   = (const float*)d_in[3];
  const float* conv_w = (const float*)d_in[4];
  const float* conv_b = (const float*)d_in[5];
  const float* W_xprj = (const float*)d_in[6];
  const float* W_dt   = (const float*)d_in[7];
  const float* b_dt   = (const float*)d_in[8];
  const float* A_log  = (const float*)d_in[9];
  const float* Dvec   = (const float*)d_in[10];
  const float* W_out  = (const float*)d_in[11];
  float* out = (float*)d_out;

  char* ws = (char*)d_ws;
  size_t off = 0;
  auto carve = [&](size_t bytes) -> char* {
    char* p = ws + off;
    off = (off + bytes + 255) & ~(size_t)255;
    return p;
  };
  _Float16* hln   = (_Float16*)carve((size_t)NROWS * D_MODEL * 2);
  _Float16* WtIn  = (_Float16*)carve((size_t)NXZ * D_MODEL * 2);
  _Float16* WtXp  = (_Float16*)carve((size_t)NXP * D_INNER * 2);
  _Float16* WtDt  = (_Float16*)carve((size_t)D_INNER * DT_RANK * 2);
  _Float16* WtOut = (_Float16*)carve((size_t)D_MODEL * D_INNER * 2);
  float*    xz    = (float*)   carve((size_t)NROWS * NXZ * 4);
  float*    u32   = (float*)   carve((size_t)NROWS * D_INNER * 4);
  _Float16* u16   = (_Float16*)carve((size_t)NROWS * D_INNER * 2);
  float*    xdbl  = (float*)   carve((size_t)NROWS * NXP * 4);
  _Float16* dtp   = (_Float16*)carve((size_t)NROWS * DT_RANK * 2);
  float*    dtf   = (float*)   carve((size_t)NROWS * D_INNER * 4);
  _Float16* y16   = (_Float16*)carve((size_t)NROWS * D_INNER * 2);

  // 1. LayerNorm -> f16
  k_layernorm<<<NROWS, 256, 0, stream>>>(x, ln_g, ln_b, hln);

  // 2. weight transpose+convert to f16 [N,K]
  k_transpose_f16<<<(D_MODEL * NXZ) / 256, 256, 0, stream>>>(W_in,  WtIn,  D_MODEL, NXZ);
  k_transpose_f16<<<(D_INNER * NXP) / 256, 256, 0, stream>>>(W_xprj, WtXp, D_INNER, NXP);
  k_transpose_f16<<<(DT_RANK * D_INNER) / 256, 256, 0, stream>>>(W_dt, WtDt, DT_RANK, D_INNER);
  k_transpose_f16<<<(D_INNER * D_MODEL) / 256, 256, 0, stream>>>(W_out, WtOut, D_INNER, D_MODEL);

  // 3. in_proj GEMM: [4096,1024] x [1024,4096] -> xz
  k_gemm_wmma<<<dim3(NXZ / 128, NROWS / 64), 256, 0, stream>>>(
      hln, WtIn, xz, NROWS, NXZ, D_MODEL, 0, nullptr, nullptr);

  // 4. causal depthwise conv + SiLU -> u (f32 + f16)
  k_conv_silu<<<(NROWS * D_INNER) / 256, 256, 0, stream>>>(xz, conv_w, conv_b, u32, u16);

  // 5. x_proj GEMM: [4096,2048] x [2048,96] -> xdbl (N=96: waves past col 96 exit)
  k_gemm_wmma<<<dim3((NXP + 127) / 128, NROWS / 64), 256, 0, stream>>>(
      u16, WtXp, xdbl, NROWS, NXP, D_INNER, 0, nullptr, nullptr);

  // 6. dt-rank slice -> f16
  k_slice_f16<<<(NROWS * DT_RANK) / 256, 256, 0, stream>>>(xdbl, dtp);

  // 7. dt_proj GEMM + softplus(acc + b_dt): [4096,64] x [64,2048] -> dtf
  k_gemm_wmma<<<dim3(D_INNER / 128, NROWS / 64), 256, 0, stream>>>(
      dtp, WtDt, dtf, NROWS, D_INNER, DT_RANK, 1, b_dt, nullptr);

  // 8. selective scan + skip + gate -> y (f16)
  k_scan<<<dim3(D_INNER / 256, B_SZ), 256, 0, stream>>>(
      dtf, u32, xdbl, xz, A_log, Dvec, y16);

  // 9. out_proj GEMM + residual: [4096,2048] x [2048,1024] + x -> out
  k_gemm_wmma<<<dim3(D_MODEL / 128, NROWS / 64), 256, 0, stream>>>(
      y16, WtOut, out, NROWS, D_MODEL, D_INNER, 2, nullptr, x);
}
